// DPSN_48515950576548
// MI455X (gfx1250) — compile-verified, matrix operand order checked by reading, not run on hardware
//
#include <hip/hip_runtime.h>
#include <stdint.h>

// Problem constants (match reference)
#define D      512
#define POOLN  65536
#define HID    256
#define CHID   128
#define KMAX   256
#define KMIN   100
#define NTOK   2048
#define CHUNK  256                 // score rows materialized at a time (64 MB f32)
#define NCHUNK (NTOK / CHUNK)

// ---- vector types (clang ext vectors; unions stay trivially constructible) ----
typedef __attribute__((ext_vector_type(16))) __bf16   v16bf;
typedef __attribute__((ext_vector_type(8)))  float    v8f;
typedef __attribute__((ext_vector_type(4)))  unsigned u32x4;
typedef __attribute__((ext_vector_type(2)))  unsigned u32x2;
typedef __attribute__((ext_vector_type(4)))  float    f32x4;

union FragBF { u32x4 u[2]; v16bf v; };

__device__ __forceinline__ unsigned short f2bf(float f) {   // round-to-nearest-even
  unsigned u = __float_as_uint(f);
  u += 0x7FFFu + ((u >> 16) & 1u);
  return (unsigned short)(u >> 16);
}
__device__ __forceinline__ unsigned monoKey(float f) {      // order-preserving f32->u32
  unsigned u = __float_as_uint(f);
  return u ^ (unsigned)(((int)u >> 31) | 0x80000000);
}
__device__ __forceinline__ float unmonoKey(unsigned k) {
  unsigned m = (unsigned)(((int)(~k) >> 31) | 0x80000000);
  return __uint_as_float(k ^ m);
}

// ---------------- kernel: Ws2 f32 -> bf16 ----------------
__global__ void k_w2_to_bf16(const float* __restrict__ w2,
                             unsigned short* __restrict__ w2b, int n4) {
  int i = blockIdx.x * blockDim.x + threadIdx.x;
  if (i >= n4) return;
  f32x4 v = *(const f32x4*)(const void*)(w2 + (size_t)i * 4);
  u32x2 o;
  o.x = (unsigned)f2bf(v.x) | ((unsigned)f2bf(v.y) << 16);
  o.y = (unsigned)f2bf(v.z) | ((unsigned)f2bf(v.w) << 16);
  *(u32x2*)(void*)(w2b + (size_t)i * 4) = o;
}

// ---------------- kernel: complexity net -> per-token budget ----------------
__global__ void __launch_bounds__(CHID) k_budget(const float* __restrict__ x,
    const float* __restrict__ Wc1, const float* __restrict__ bc1,
    const float* __restrict__ Wc2, const float* __restrict__ bc2,
    int* __restrict__ budget) {
  __shared__ float xs[D];
  __shared__ float red[CHID];
  const int row = blockIdx.x, t = threadIdx.x;
  for (int i = t; i < D; i += CHID) xs[i] = x[(size_t)row * D + i];
  __syncthreads();
  float acc = bc1[t];
  for (int k = 0; k < D; ++k) acc = fmaf(xs[k], Wc1[k * CHID + t], acc);
  red[t] = fmaxf(acc, 0.0f) * Wc2[t];
  __syncthreads();
  for (int s = CHID / 2; s > 0; s >>= 1) {
    if (t < s) red[t] += red[t + s];
    __syncthreads();
  }
  if (t == 0) {
    float z = red[0] + bc2[0];
    float c = 1.0f / (1.0f + expf(-z));          // sigmoid
    float raw = (float)KMIN + (float)(KMAX - KMIN) * c * c;  // complexity^2
    raw = fminf(fmaxf(raw, (float)KMIN), (float)KMAX);
    budget[row] = (int)rintf(raw);               // ties-to-even like jnp.round
  }
}

// ---------------- kernel: h = relu(x@Ws1 + bs1), stored bf16 ----------------
__global__ void __launch_bounds__(HID) k_hidden(const float* __restrict__ x,
    const float* __restrict__ Ws1, const float* __restrict__ bs1,
    unsigned short* __restrict__ hb) {
  __shared__ float xs[D];
  const int row = blockIdx.x, t = threadIdx.x;
  for (int i = t; i < D; i += HID) xs[i] = x[(size_t)row * D + i];
  __syncthreads();
  float acc = bs1[t];
  for (int k = 0; k < D; ++k) acc = fmaf(xs[k], Ws1[k * HID + t], acc);
  hb[(size_t)row * HID + t] = f2bf(fmaxf(acc, 0.0f));
}

// ---------------- kernel: scores chunk = h @ Ws2 + bs2  (bf16 WMMA) ----------------
// block = 256 threads = 8 wave32s; tile M=64 x N=128; K=256 in 8 chunks of 32.
// Each wave: 4 M-subtiles x 1 N-subtile -> 4 accumulators, 32 WMMAs total.
// B tile double-buffered in LDS (transposed at store: row n, 80B pitch).
// A fragments are batched into independent registers ahead of the WMMA block so
// VMEM latency overlaps the matrix pipe instead of forcing per-WMMA full waits.
__global__ void __launch_bounds__(256) k_scores_wmma(
    const unsigned short* __restrict__ hb, const unsigned* __restrict__ w2b32,
    const float* __restrict__ bs2, float* __restrict__ sc, int chunk) {
  __attribute__((aligned(16))) __shared__ unsigned short Bt[2][128 * 40];

  const int t    = threadIdx.x;
  const int wave = t >> 5;          // gfx1250 wave32
  const int lane = t & 31;
  const int half = lane >> 4;       // K sub-block select per 16-bit WMMA layout
  const int l16  = lane & 15;
  const int mBase = blockIdx.x * 64;            // row tile inside chunk (4 subtiles)
  const int nBase = blockIdx.y * 128;

  // per-lane A row base pointers for the 4 M-subtiles
  const unsigned short* aRow[4];
#pragma unroll
  for (int s = 0; s < 4; ++s)
    aRow[s] = hb + (size_t)(chunk * CHUNK + mBase + s * 16 + l16) * HID + half * 8;

  v8f acc[4] = {{}, {}, {}, {}};

  auto stage = [&](int kc, int b) {   // stage 32(K) x 128(N) bf16, transposed
#pragma unroll
    for (int q = 0; q < 8; ++q) {
      int li  = q * 256 + t;          // 2048 dwords = 32 k-rows x 64 dword-cols
      int kr  = li >> 6;
      int nc2 = li & 63;
      unsigned w = w2b32[(size_t)(kc * 32 + kr) * (POOLN / 2) + (nBase >> 1) + nc2];
      int n0 = nc2 * 2;
      Bt[b][n0 * 40 + kr]       = (unsigned short)(w & 0xFFFFu);
      Bt[b][(n0 + 1) * 40 + kr] = (unsigned short)(w >> 16);
    }
  };

  stage(0, 0);
#pragma unroll
  for (int kc = 0; kc < 8; ++kc) {
    __syncthreads();                             // buffer kc&1 ready for reads
    if (kc < 7) {
      __builtin_prefetch(&w2b32[(size_t)((kc + 1) * 32) * (POOLN / 2) + (nBase >> 1)], 0, 3);
      stage(kc + 1, (kc + 1) & 1);               // fill other buffer while computing
    }

    // issue all VMEM A-fragment loads first (independent destinations) ...
    FragBF fa[4];
#pragma unroll
    for (int s = 0; s < 4; ++s) {
      const unsigned short* aP = aRow[s] + kc * 32;
      fa[s].u[0] = *(const u32x4*)(const void*)(aP);
      fa[s].u[1] = *(const u32x4*)(const void*)(aP + 16);
    }
    // ... then the LDS B-fragment loads ...
    FragBF fb;
    const unsigned short* bP = &Bt[kc & 1][(wave * 16 + l16) * 40 + half * 8];
    fb.u[0] = *(const u32x4*)(const void*)(bP);
    fb.u[1] = *(const u32x4*)(const void*)(bP + 16);

    // ... then four back-to-back WMMAs sharing the B fragment.
#pragma unroll
    for (int s = 0; s < 4; ++s)
      acc[s] = __builtin_amdgcn_wmma_f32_16x16x32_bf16(false, fa[s].v, false, fb.v,
                                                       (short)0, acc[s], false, false);
  }

  // epilogue: C/D layout — lane n = l16, rows M = half*8 + r in VGPR r
  const int nG = nBase + wave * 16 + l16;
  const float bias = bs2[nG];
#pragma unroll
  for (int s = 0; s < 4; ++s) {
#pragma unroll
    for (int r = 0; r < 8; ++r) {
      int m = mBase + s * 16 + half * 8 + r;     // local row within chunk
      sc[(size_t)m * POOLN + nG] = acc[s][r] + bias;
    }
  }
}

// ---------------- kernel: per-row top-256 via radix select + bitonic sort ----------------
__global__ void __launch_bounds__(256) k_select(
    const float* __restrict__ sc, const int* __restrict__ budget,
    int* __restrict__ tidx, float* __restrict__ tw, int chunk) {
  __shared__ unsigned hist[2048];
  __shared__ unsigned selKey[KMAX];
  __shared__ unsigned selIdx[KMAX];
  __shared__ unsigned suf[256];
  __shared__ float    red[256];
  __shared__ unsigned sPrefix;
  __shared__ int      sKrem;
  __shared__ unsigned sCntGt, sCntEq;

  const int t    = threadIdx.x;
  const int rowL = blockIdx.x;
  const int rowG = chunk * CHUNK + rowL;
  const float* row = sc + (size_t)rowL * POOLN;

  if (t == 0) { sKrem = KMAX; sPrefix = 0u; sCntGt = 0u; sCntEq = 0u; }
  selKey[t] = 0u; selIdx[t] = 0u;

  const int shifts[3] = {21, 10, 0};
  const int nbins[3]  = {2048, 2048, 1024};
  for (int lvl = 0; lvl < 3; ++lvl) {
    const int nb = nbins[lvl];
    for (int i = t; i < 2048; i += 256) hist[i] = 0u;
    __syncthreads();
    const unsigned pref = sPrefix;
    const int sh = shifts[lvl];
    const int matchShift = (lvl == 0) ? 0 : shifts[lvl - 1];
    for (int i = t; i < POOLN; i += 256) {
      unsigned key = monoKey(row[i]);
      if (lvl == 0 || (key >> matchShift) == pref)
        atomicAdd(&hist[(key >> sh) & (unsigned)(nb - 1)], 1u);
    }
    __syncthreads();

    // parallel threshold-bin selection: per-thread totals + suffix scan
    const int bpt = nb >> 8;                 // bins per thread (8 or 4)
    const int b0  = t * bpt;
    unsigned localTot = 0;
    for (int b = 0; b < bpt; ++b) localTot += hist[b0 + b];
    suf[t] = localTot;
    __syncthreads();
    unsigned run = localTot;                 // becomes inclusive suffix sum t..255
    for (int off = 1; off < 256; off <<= 1) {
      unsigned add = (t + off < 256) ? suf[t + off] : 0u;
      __syncthreads();
      run += add;
      suf[t] = run;
      __syncthreads();
    }
    const int krem = sKrem;
    const unsigned above = run - localTot;   // count in bins strictly above this thread's
    if (above < (unsigned)krem && run >= (unsigned)krem) {   // unique owner thread
      unsigned cum = above; int chosen = b0;
      for (int b = b0 + bpt - 1; b >= b0; --b) {
        cum += hist[b];
        if (cum >= (unsigned)krem) { chosen = b; break; }
      }
      sKrem   = krem - (int)(cum - hist[chosen]);
      sPrefix = (pref << ((lvl == 2) ? 10 : 11)) | (unsigned)chosen;
    }
    __syncthreads();
  }

  const unsigned T  = sPrefix;       // exact 256th-largest key
  const int ties    = sKrem;         // how many ==T to take
  const int nGt     = KMAX - ties;

  for (int i = t; i < POOLN; i += 256) {
    unsigned key = monoKey(row[i]);
    if (key > T) {
      unsigned p = atomicAdd(&sCntGt, 1u);
      if (p < (unsigned)nGt) { selKey[p] = key; selIdx[p] = (unsigned)i; }
    } else if (key == T) {
      unsigned e = atomicAdd(&sCntEq, 1u);
      if (e < (unsigned)ties) { selKey[nGt + e] = key; selIdx[nGt + e] = (unsigned)i; }
    }
  }
  __syncthreads();

  // bitonic sort 256 pairs, descending by value, ascending by index on ties
  for (int k = 2; k <= KMAX; k <<= 1) {
    for (int j = k >> 1; j > 0; j >>= 1) {
      int i = t, p = i ^ j;
      if (p > i) {
        unsigned ka = selKey[i], kb = selKey[p];
        unsigned ia = selIdx[i], ib = selIdx[p];
        bool aFirst = (ka > kb) || (ka == kb && ia < ib);
        bool up = ((i & k) == 0);
        if (aFirst != up) {
          selKey[i] = kb; selKey[p] = ka;
          selIdx[i] = ib; selIdx[p] = ia;
        }
      }
      __syncthreads();
    }
  }

  // softmax over all 256 sorted values, then mask positions >= budget
  float vmax = unmonoKey(selKey[0]);
  float v = unmonoKey(selKey[t]);
  float e = expf(v - vmax);
  red[t] = e;
  __syncthreads();
  for (int s = 128; s > 0; s >>= 1) {
    if (t < s) red[t] += red[t + s];
    __syncthreads();
  }
  const float denom = red[0];
  int bud = budget[rowG];
  tidx[(size_t)rowG * KMAX + t] = (int)selIdx[t];
  tw  [(size_t)rowG * KMAX + t] = (t < bud) ? (e / denom) : 0.0f;
}

// ---------------- kernel: executor (gather + tanh-gated rank-1 accumulate + residual) ----------------
__global__ void __launch_bounds__(256) k_exec(
    const float* __restrict__ x, const float* __restrict__ pool,
    const int* __restrict__ budget, const int* __restrict__ tidx,
    const float* __restrict__ tw, float* __restrict__ out) {
  __shared__ float xs[D];
  __shared__ float acts[KMAX];
  __shared__ int   idxs[KMAX];
  const int row = blockIdx.x, t = threadIdx.x;
  xs[t]       = x[(size_t)row * D + t];
  xs[t + 256] = x[(size_t)row * D + t + 256];
  idxs[t] = tidx[(size_t)row * KMAX + t];
  acts[t] = 0.0f;
  __syncthreads();

  int bud = budget[row];
  if (bud > KMAX) bud = KMAX;
  const int wave = t >> 5, lane = t & 31;   // wave32

  // phase 1: acts[k] = tanh(x . pool[idx_k]) * w_k for live (k < budget) entries
  for (int k = wave; k < bud; k += 8) {
    const float* pr = pool + (size_t)idxs[k] * D;
    float partial = 0.0f;
    for (int d = lane; d < D; d += 32) partial = fmaf(xs[d], pr[d], partial);
#pragma unroll
    for (int off = 16; off > 0; off >>= 1) partial += __shfl_xor(partial, off, 32);
    if (lane == 0) acts[k] = tanhf(partial) * tw[(size_t)row * KMAX + k];
  }
  __syncthreads();

  // phase 2: out = sum_k acts[k] * pool[idx_k] + x
  float a0 = xs[t], a1 = xs[t + 256];
  for (int k = 0; k < bud; ++k) {
    float a = acts[k];
    const float* pr = pool + (size_t)idxs[k] * D;
    a0 = fmaf(a, pr[t], a0);
    a1 = fmaf(a, pr[t + 256], a1);
  }
  out[(size_t)row * D + t]       = a0;
  out[(size_t)row * D + t + 256] = a1;
}

// ---------------- launch ----------------
extern "C" void kernel_launch(void* const* d_in, const int* in_sizes, int n_in,
                              void* d_out, int out_size, void* d_ws, size_t ws_size,
                              hipStream_t stream) {
  (void)in_sizes; (void)n_in; (void)out_size; (void)ws_size;
  const float* x    = (const float*)d_in[0];
  const float* pool = (const float*)d_in[1];
  const float* Wc1  = (const float*)d_in[2];
  const float* bc1  = (const float*)d_in[3];
  const float* Wc2  = (const float*)d_in[4];
  const float* bc2  = (const float*)d_in[5];
  const float* Ws1  = (const float*)d_in[6];
  const float* bs1  = (const float*)d_in[7];
  const float* Ws2  = (const float*)d_in[8];
  const float* bs2  = (const float*)d_in[9];
  float* out = (float*)d_out;

  // scratch layout (~106 MB total; all sections 256B-aligned)
  char* p = (char*)d_ws;
  unsigned short* w2b = (unsigned short*)p;  p += (size_t)HID * POOLN * 2;   // 33.5 MB
  unsigned short* hb  = (unsigned short*)p;  p += (size_t)NTOK * HID * 2;    //  1.0 MB
  int*   budget       = (int*)p;             p += (size_t)NTOK * 4;          //  8 KB
  int*   tidx         = (int*)p;             p += (size_t)NTOK * KMAX * 4;   //  2.1 MB
  float* tw           = (float*)p;           p += (size_t)NTOK * KMAX * 4;   //  2.1 MB
  float* sc           = (float*)p;                                           // 64 MB chunk

  const int n4 = HID * POOLN / 4;
  k_w2_to_bf16<<<dim3((n4 + 255) / 256), dim3(256), 0, stream>>>(Ws2, w2b, n4);
  k_budget<<<dim3(NTOK), dim3(CHID), 0, stream>>>(x, Wc1, bc1, Wc2, bc2, budget);
  k_hidden<<<dim3(NTOK), dim3(HID), 0, stream>>>(x, Ws1, bs1, hb);

  for (int c = 0; c < NCHUNK; ++c) {
    k_scores_wmma<<<dim3(CHUNK / 64, POOLN / 128), dim3(256), 0, stream>>>(
        hb, (const unsigned*)w2b, bs2, sc, c);
    k_select<<<dim3(CHUNK), dim3(256), 0, stream>>>(sc, budget, tidx, tw, c);
  }

  k_exec<<<dim3(NTOK), dim3(256), 0, stream>>>(x, pool, budget, tidx, tw, out);
}
